// Model_84198538871259
// MI455X (gfx1250) — matrix operand order repaired and checked by reading, not compiled, not run on hardware
//
#include <hip/hip_runtime.h>

typedef __attribute__((ext_vector_type(16))) _Float16 v16h;
typedef __attribute__((ext_vector_type(2)))  _Float16 v2h;
typedef __attribute__((ext_vector_type(8)))  float    v8f;

#define DEVFN __device__ __forceinline__

constexpr int BB    = 8;    // batch
constexpr int NNODE = 512;  // nodes
constexpr int ICH   = 16;   // i-chunks per (b, j-group)

struct KnetArgs { const float* w[9]; const float* b[9]; };
struct HnetArgs { const float* w[6]; const float* b[6]; };
struct HeadArgs { const float* w[5]; const float* b[5]; };

// D = A(16x32 f16) * B(32x16 f16) + C(16x16 f32)
DEVFN v8f wmma16(v16h a, v16h b, v8f c) {
  return __builtin_amdgcn_wmma_f32_16x16x32_f16(false, a, false, b, (short)0, c,
                                                false, false);
}

// v_cvt_pk_rtz_f16_f32: two f32 -> packed f16 pair (bit-cast to _Float16 vec).
DEVFN v2h cvt_pk(float a, float b) {
  auto p = __builtin_amdgcn_cvt_pkrtz(a, b);   // __fp16 ext_vector_type(2)
  return __builtin_bit_cast(v2h, p);
}

// Packed-f16 ReLU in exactly one instruction (avoids clang's canonicalize+max pair).
DEVFN v2h relu_pk(v2h x) {
  unsigned u = __builtin_bit_cast(unsigned, x), r;
  asm("v_pk_max_num_f16 %0, %1, 0" : "=v"(r) : "v"(u));
  return __builtin_bit_cast(v2h, r);
}

// Augmented A operand: [W | b] with bias at K-row 'kin' (one past the weights).
// Element e of lane: K = e + hi*16 ; M = col ; rows m>=8 zero.
DEVFN v16h load_wA(const float* w, const float* bias, int kin, int col, int hi) {
  v16h a;
#pragma unroll
  for (int e = 0; e < 16; ++e) {
    int k = e + hi * 16;
    float v = 0.0f;
    if (col < 8) {
      if (k < kin)       v = w[k * 8 + col];
      else if (k == kin) v = bias[col];       // bias row (pairs with B-row == 1.0)
    }
    a[e] = (_Float16)v;
  }
  return a;
}

// D (f32 C-layout, bias already included via augmented A) -> next-layer B operand.
// Only halves 0..7 (K-rows 0..7) are rewritten: K-rows 9..31 multiply zero
// A-columns, so stale data there is inert.  Half 8 (K-row 8) carries the
// constant 1.0 that pairs with the bias column of the next layer; set_one
// re-establishes it when the previous B operand held real data there.
DEVFN void pack_act(v16h& r, v8f d, bool relu, bool set_one = false) {
#pragma unroll
  for (int g = 0; g < 4; ++g) {
    v2h p = cvt_pk(d[2 * g], d[2 * g + 1]);
    if (relu) p = relu_pk(p);
    r[2 * g] = p[0]; r[2 * g + 1] = p[1];
  }
  if (set_one) r[8] = (_Float16)1.0f;
}

__global__ void zero_kernel(float* __restrict__ p, int n) {
  int i = blockIdx.x * blockDim.x + threadIdx.x;
  if (i < n) p[i] = 0.0f;
}

// One wave handles TWO 16-j tiles (32 j-columns) for one batch, iterating an
// i-chunk: two independent WMMA chains share the A operands and fill each
// other's WMMA->VALU hazard slots.
// msg[b,j,:] += sum_i knet(concat(r_j, r_i, e_ij) * A_ij)   (atomic across chunks)
template <int CIN>
__global__ __launch_bounds__(32, 1)
void conv_edge_kernel(const float* __restrict__ rin, const float* __restrict__ xyz,
                      KnetArgs P, float* __restrict__ msg) {
  constexpr int KIN0  = 2 * CIN + 1;   // layer-0 width; bias row = KIN0
  constexpr int CHUNK = NNODE / ICH;
  const int lane = threadIdx.x;
  const int col  = lane & 15;
  const int hi   = lane >> 4;

  unsigned blk = blockIdx.x;
  const int jt = blk % (NNODE / 32); blk /= (NNODE / 32);
  const int b  = blk % BB;           blk /= BB;
  const int ic = blk;
  const int j0 = jt * 32 + col;
  const int j1 = j0 + 16;
  const int i0 = ic * CHUNK;

  v16h A[9];
#pragma unroll
  for (int L = 0; L < 9; ++L)
    A[L] = load_wA(P.w[L], P.b[L], (L == 0) ? KIN0 : 8, col, hi);

  const float xj00 = xyz[(b * NNODE + j0) * 3 + 0];
  const float xj01 = xyz[(b * NNODE + j0) * 3 + 1];
  const float xj02 = xyz[(b * NNODE + j0) * 3 + 2];
  const float xj10 = xyz[(b * NNODE + j1) * 3 + 0];
  const float xj11 = xyz[(b * NNODE + j1) * 3 + 1];
  const float xj12 = xyz[(b * NNODE + j1) * 3 + 2];
  float rj0[CIN], rj1[CIN];
#pragma unroll
  for (int k = 0; k < CIN; ++k) {
    rj0[k] = rin[(b * NNODE + j0) * CIN + k];
    rj1[k] = rin[(b * NNODE + j1) * CIN + k];
  }

  v8f macc0, macc1, zc;
#pragma unroll
  for (int g = 0; g < 8; ++g) { macc0[g] = 0.0f; macc1[g] = 0.0f; zc[g] = 0.0f; }

  // Software-pipelined node data (uniform loads issued one iteration ahead).
  float xi0 = xyz[(b * NNODE + i0) * 3 + 0];
  float xi1 = xyz[(b * NNODE + i0) * 3 + 1];
  float xi2 = xyz[(b * NNODE + i0) * 3 + 2];
  float ri[CIN];
#pragma unroll
  for (int k = 0; k < CIN; ++k) ri[k] = rin[(b * NNODE + i0) * CIN + k];

  v16h Bm0, Bm1;
#pragma unroll 1
  for (int i = i0; i < i0 + CHUNK; ++i) {
    const float cxi0 = xi0, cxi1 = xi1, cxi2 = xi2;
    float cri[CIN];
#pragma unroll
    for (int k = 0; k < CIN; ++k) cri[k] = ri[k];

    const int inx = (i + 1 < NNODE) ? i + 1 : i;   // prefetch next node
    xi0 = xyz[(b * NNODE + inx) * 3 + 0];
    xi1 = xyz[(b * NNODE + inx) * 3 + 1];
    xi2 = xyz[(b * NNODE + inx) * 3 + 2];
#pragma unroll
    for (int k = 0; k < CIN; ++k) ri[k] = rin[(b * NNODE + inx) * CIN + k];

    const float dx0 = cxi0 - xj00, dy0 = cxi1 - xj01, dz0 = cxi2 - xj02;
    const float ds0 = dx0 * dx0 + dy0 * dy0 + dz0 * dz0;
    const float aj0 = (ds0 <= 4.0f && ds0 != 0.0f) ? 1.0f : 0.0f;
    const float ev0 = ds0 * aj0;
    const float dx1 = cxi0 - xj10, dy1 = cxi1 - xj11, dz1 = cxi2 - xj12;
    const float ds1 = dx1 * dx1 + dy1 * dy1 + dz1 * dz1;
    const float aj1 = (ds1 <= 4.0f && ds1 != 0.0f) ? 1.0f : 0.0f;
    const float ev1 = ds1 * aj1;

    // B operand. Lo lanes: K=0..15. Hi lanes: K=16..31 (only K=16/17 matter).
    auto buildB = [&](v16h& Bm, const float* rjv, float adj, float ev) {
      float t[16];
#pragma unroll
      for (int e = 0; e < 16; ++e) {
        float v = 0.0f;
        if (e < CIN)                v = rjv[e] * adj;
        else if (e < 2 * CIN)       v = cri[e - CIN] * adj;
        else if (e == 2 * CIN)      v = ev;
        else if (e == 2 * CIN + 1)  v = 1.0f;   // layer-0 bias row (CIN==1: K=3)
        if (CIN == 1 && e == 8)     v = 1.0f;   // bias row for layers >= 1
        t[e] = v;
      }
      if (CIN == 8) {
        t[0] = hi ? ev   : t[0];   // K=16: edge feature
        t[1] = hi ? 1.0f : t[1];   // K=17: layer-0 bias row
      }
#pragma unroll
      for (int g = 0; g < 8; ++g) {
        v2h p = cvt_pk(t[2 * g], t[2 * g + 1]);
        Bm[2 * g] = p[0]; Bm[2 * g + 1] = p[1];
      }
    };
    buildB(Bm0, rj0, aj0, ev0);
    buildB(Bm1, rj1, aj1, ev1);

    v8f d0, d1;
    d0 = wmma16(A[0], Bm0, zc); d1 = wmma16(A[0], Bm1, zc);
    pack_act(Bm0, d0, false, CIN == 8); pack_act(Bm1, d1, false, CIN == 8); // k0
    d0 = wmma16(A[1], Bm0, zc); d1 = wmma16(A[1], Bm1, zc);
    pack_act(Bm0, d0, false); pack_act(Bm1, d1, false);                     // k1
    d0 = wmma16(A[2], Bm0, zc); d1 = wmma16(A[2], Bm1, zc);
    pack_act(Bm0, d0, true);  pack_act(Bm1, d1, true);                      // k2.l1
    d0 = wmma16(A[3], Bm0, zc); d1 = wmma16(A[3], Bm1, zc);
    pack_act(Bm0, d0, true);  pack_act(Bm1, d1, true);                      // k2.l2
    d0 = wmma16(A[4], Bm0, zc); d1 = wmma16(A[4], Bm1, zc);
    pack_act(Bm0, d0, true);  pack_act(Bm1, d1, true);                      // k3.l1
    d0 = wmma16(A[5], Bm0, zc); d1 = wmma16(A[5], Bm1, zc);
    pack_act(Bm0, d0, true);  pack_act(Bm1, d1, true);                      // k3.l2
    d0 = wmma16(A[6], Bm0, zc); d1 = wmma16(A[6], Bm1, zc);
    pack_act(Bm0, d0, true);  pack_act(Bm1, d1, true);                      // k4
    d0 = wmma16(A[7], Bm0, zc); d1 = wmma16(A[7], Bm1, zc);
    pack_act(Bm0, d0, true);  pack_act(Bm1, d1, true);                      // k5
    macc0 = wmma16(A[8], Bm0, macc0);                                       // k6
    macc1 = wmma16(A[8], Bm1, macc1);
  }

  if (hi == 0) {
#pragma unroll
    for (int g = 0; g < 8; ++g) {
      atomicAdd(&msg[(b * NNODE + j0) * 8 + g], macc0[g]);
      atomicAdd(&msg[(b * NNODE + j1) * 8 + g], macc1[g]);
    }
  }
}

// hnet over 16 rows per wave: rout = h3(lb(h2, lb(h1, relu(h0(msg)))))
__global__ __launch_bounds__(32, 1)
void hnet_kernel(const float* __restrict__ msg, HnetArgs P, float* __restrict__ rout) {
  const int lane = threadIdx.x;
  const int col  = lane & 15;
  const int hi   = lane >> 4;
  const int row  = blockIdx.x * 16 + col;

  v16h A[6];
#pragma unroll
  for (int L = 0; L < 6; ++L) A[L] = load_wA(P.w[L], P.b[L], 8, col, hi);

  v8f zc;
#pragma unroll
  for (int g = 0; g < 8; ++g) zc[g] = 0.0f;

  v16h Bm;
#pragma unroll
  for (int g = 0; g < 8; ++g) {
    float a = (2 * g < 8) ? msg[row * 8 + 2 * g] : (2 * g == 8 ? 1.0f : 0.0f);
    float b = (2 * g + 1 < 8) ? msg[row * 8 + 2 * g + 1] : 0.0f;
    v2h p = cvt_pk(a, b);
    Bm[2 * g] = p[0]; Bm[2 * g + 1] = p[1];
  }

  v8f d;
  d = wmma16(A[0], Bm, zc); pack_act(Bm, d, true); // h0 + relu
  d = wmma16(A[1], Bm, zc); pack_act(Bm, d, true); // h1.l1
  d = wmma16(A[2], Bm, zc); pack_act(Bm, d, true); // h1.l2
  d = wmma16(A[3], Bm, zc); pack_act(Bm, d, true); // h2.l1
  d = wmma16(A[4], Bm, zc); pack_act(Bm, d, true); // h2.l2
  d = wmma16(A[5], Bm, zc);                        // h3 (bias in A, no relu)

  if (hi == 0) {
#pragma unroll
    for (int g = 0; g < 8; ++g) rout[row * 8 + g] = d[g];
  }
}

// f[b,:] (+)= sum_n relu(x[b,n,:] @ W + bias)
template <int CIN, bool ACC>
__global__ __launch_bounds__(64)
void fp_kernel(const float* __restrict__ x, const float* __restrict__ w,
               const float* __restrict__ bias, float* __restrict__ f) {
  const int b = blockIdx.x;
  const int o = threadIdx.x;
  float wl[CIN];
#pragma unroll
  for (int k = 0; k < CIN; ++k) wl[k] = w[k * 64 + o];
  const float bo = bias[o];
  float acc = 0.0f;
  for (int i = 0; i < NNODE; ++i) {
    float s = bo;
#pragma unroll
    for (int k = 0; k < CIN; ++k) s += x[(b * NNODE + i) * CIN + k] * wl[k];
    acc += fmaxf(s, 0.0f);
  }
  if (ACC) f[b * 64 + o] += acc;
  else     f[b * 64 + o] = acc;
}

// Final MLP: 64 -> 32 -> lb(32) -> 16(relu) -> 1
__global__ __launch_bounds__(256)
void head_kernel(const float* __restrict__ f, HeadArgs P, float* __restrict__ out) {
  __shared__ float sf[BB * 64];
  __shared__ float s0[BB * 32];
  __shared__ float s1[BB * 32];
  const int tid = threadIdx.x;
  for (int t = tid; t < BB * 64; t += 256) sf[t] = f[t];
  __syncthreads();
  const int b = tid >> 5;
  const int c = tid & 31;
  {
    float v = P.b[0][c];
    for (int k = 0; k < 64; ++k) v += sf[b * 64 + k] * P.w[0][k * 32 + c];
    s0[b * 32 + c] = v; // n0: no activation
  }
  __syncthreads();
  {
    float v = P.b[1][c];
    for (int k = 0; k < 32; ++k) v += s0[b * 32 + k] * P.w[1][k * 32 + c];
    s1[b * 32 + c] = fmaxf(v, 0.0f); // n1.l1
  }
  __syncthreads();
  {
    float v = P.b[2][c];
    for (int k = 0; k < 32; ++k) v += s1[b * 32 + k] * P.w[2][k * 32 + c];
    s0[b * 32 + c] = fmaxf(v, 0.0f); // n1.l2
  }
  __syncthreads();
  if (c < 16) {
    float v = P.b[3][c];
    for (int k = 0; k < 32; ++k) v += s0[b * 32 + k] * P.w[3][k * 16 + c];
    s1[b * 32 + c] = fmaxf(v, 0.0f); // n2 + relu
  }
  __syncthreads();
  if (c == 0) {
    float v = P.b[4][0];
    for (int k = 0; k < 16; ++k) v += s1[b * 32 + k] * P.w[4][k];
    out[b] = v; // n3
  }
}

extern "C" void kernel_launch(void* const* d_in, const int* in_sizes, int n_in,
                              void* d_out, int out_size, void* d_ws, size_t ws_size,
                              hipStream_t stream) {
  (void)out_size; (void)ws_size;
  struct Lin { const float* w; const float* b; };
  Lin fp[4]; Lin K[3][9]; Lin H[3][6]; Lin NNp[5];
  const float* r = nullptr; const float* xyz = nullptr;
  if (n_in < 3) return;

  if (n_in >= 110) {
    if (in_sizes[0] == BB * NNODE) {
      // dict insertion order: r, xyz, fp1..4(w,b), conv{K:k0..k6(lblocks l1,l2), H:h0..h3}, nn
      r = (const float*)d_in[0]; xyz = (const float*)d_in[1];
      int idx = 2;
      auto nx = [&]() { Lin l; l.w = (const float*)d_in[idx]; l.b = (const float*)d_in[idx + 1]; idx += 2; return l; };
      for (int t = 0; t < 4; ++t) fp[t] = nx();
      for (int c = 0; c < 3; ++c) {
        for (int L = 0; L < 9; ++L) K[c][L] = nx();
        for (int L = 0; L < 6; ++L) H[c][L] = nx();
      }
      for (int L = 0; L < 5; ++L) NNp[L] = nx();
    } else {
      // jax sorted pytree: conv1..3 (H before K, b before w), fp1..4, nn, then r, xyz
      int idx = 0;
      auto nx = [&]() { Lin l; l.b = (const float*)d_in[idx]; l.w = (const float*)d_in[idx + 1]; idx += 2; return l; };
      for (int c = 0; c < 3; ++c) {
        for (int L = 0; L < 6; ++L) H[c][L] = nx();
        for (int L = 0; L < 9; ++L) K[c][L] = nx();
      }
      for (int t = 0; t < 4; ++t) fp[t] = nx();
      for (int L = 0; L < 5; ++L) NNp[L] = nx();
      r = (const float*)d_in[idx]; xyz = (const float*)d_in[idx + 1];
    }
  } else {
    // params as one concatenated flat buffer (sorted leaf order) in d_in[2]
    r = (const float*)d_in[0]; xyz = (const float*)d_in[1];
    const float* base = (const float*)d_in[2];
    size_t off = 0;
    auto nx = [&](int fin, int fout) { Lin l; l.b = base + off; off += fout;
                                       l.w = base + off; off += (size_t)fin * fout; return l; };
    for (int c = 0; c < 3; ++c) {
      int cin = (c == 0) ? 1 : 8;
      for (int L = 0; L < 6; ++L) H[c][L] = nx(8, 8);
      K[c][0] = nx(2 * cin + 1, 8);
      for (int L = 1; L < 9; ++L) K[c][L] = nx(8, 8);
    }
    fp[0] = nx(1, 64);
    for (int t = 1; t < 4; ++t) fp[t] = nx(8, 64);
    NNp[0] = nx(64, 32); NNp[1] = nx(32, 32); NNp[2] = nx(32, 32);
    NNp[3] = nx(32, 16); NNp[4] = nx(16, 1);
  }

  float* ws   = (float*)d_ws;
  float* msgb = ws;              // B*N*8 = 32768 floats
  float* r1   = ws + 32768;
  float* r2   = ws + 65536;
  float* r3   = ws + 98304;
  float* fbuf = ws + 131072;     // B*64 = 512 floats

  auto kargs = [&](int c) { KnetArgs a; for (int L = 0; L < 9; ++L) { a.w[L] = K[c][L].w; a.b[L] = K[c][L].b; } return a; };
  auto hargs = [&](int c) { HnetArgs a; for (int L = 0; L < 6; ++L) { a.w[L] = H[c][L].w; a.b[L] = H[c][L].b; } return a; };

  const int convGrid = (NNODE / 32) * BB * ICH;
  const int msgN = BB * NNODE * 8;

  fp_kernel<1, false><<<BB, 64, 0, stream>>>(r, fp[0].w, fp[0].b, fbuf);

  zero_kernel<<<(msgN + 255) / 256, 256, 0, stream>>>(msgb, msgN);
  conv_edge_kernel<1><<<convGrid, 32, 0, stream>>>(r, xyz, kargs(0), msgb);
  hnet_kernel<<<BB * NNODE / 16, 32, 0, stream>>>(msgb, hargs(0), r1);
  fp_kernel<8, true><<<BB, 64, 0, stream>>>(r1, fp[1].w, fp[1].b, fbuf);

  zero_kernel<<<(msgN + 255) / 256, 256, 0, stream>>>(msgb, msgN);
  conv_edge_kernel<8><<<convGrid, 32, 0, stream>>>(r1, xyz, kargs(1), msgb);
  hnet_kernel<<<BB * NNODE / 16, 32, 0, stream>>>(msgb, hargs(1), r2);
  fp_kernel<8, true><<<BB, 64, 0, stream>>>(r2, fp[2].w, fp[2].b, fbuf);

  zero_kernel<<<(msgN + 255) / 256, 256, 0, stream>>>(msgb, msgN);
  conv_edge_kernel<8><<<convGrid, 32, 0, stream>>>(r2, xyz, kargs(2), msgb);
  hnet_kernel<<<BB * NNODE / 16, 32, 0, stream>>>(msgb, hargs(2), r3);
  fp_kernel<8, true><<<BB, 64, 0, stream>>>(r3, fp[3].w, fp[3].b, fbuf);

  HeadArgs hd;
  for (int L = 0; L < 5; ++L) { hd.w[L] = NNp[L].w; hd.b[L] = NNp[L].b; }
  head_kernel<<<1, 256, 0, stream>>>(fbuf, hd, (float*)d_out);
}
